// PairHead17_82858509074932
// MI455X (gfx1250) — matrix operand order repaired and checked by reading, not compiled, not run on hardware
//
#include <hip/hip_runtime.h>
#include <hip/hip_bf16.h>

// ---------------------------------------------------------------------------
// Problem constants (match reference)
// ---------------------------------------------------------------------------
#define TB   8
#define TP   4096
#define TN   128
#define TT   (TB * TP)        // 32768 pair rows
#define NCLS 133

typedef __attribute__((ext_vector_type(16))) __bf16 bf16x16;
typedef __attribute__((ext_vector_type(8)))  float  f32x8;

union FragBF { bf16x16 v; unsigned int u[8]; };

static __device__ __forceinline__ unsigned short f2bf(float f) {
    union { float f; unsigned int u; } x; x.f = f;
    unsigned int r = x.u + 0x7FFFu + ((x.u >> 16) & 1u);   // round-to-nearest-even
    return (unsigned short)(r >> 16);
}

// ---------------------------------------------------------------------------
// Async global->LDS staging (CDNA5 GLOBAL_LOAD_ASYNC_TO_LDS_B128, ASYNCcnt).
// The clang-22 builtin takes (v4i AS1*, v4i AS3*, imm offset, imm cpol).
// ---------------------------------------------------------------------------
#if defined(__has_builtin)
#if __has_builtin(__builtin_amdgcn_global_load_async_to_lds_b128)
#define HAVE_ASYNC_BUILTIN 1
#endif
#if __has_builtin(__builtin_amdgcn_s_wait_asynccnt)
#define HAVE_WAIT_BUILTIN 1
#endif
#endif

typedef int v4i __attribute__((vector_size(16)));
typedef __attribute__((address_space(1))) v4i* gv4i_p;
typedef __attribute__((address_space(3))) v4i* lv4i_p;

static __device__ __forceinline__
void async_copy16(const unsigned short* g, unsigned short* l) {
#ifdef HAVE_ASYNC_BUILTIN
    __builtin_amdgcn_global_load_async_to_lds_b128((gv4i_p)g, (lv4i_p)l, 0, 0);
#else
    unsigned ldsOff = (unsigned)(size_t)(__attribute__((address_space(3))) void*)l;
    asm volatile("global_load_async_to_lds_b128 %0, %1, off"
                 :: "v"(ldsOff), "v"(g) : "memory");
#endif
}

static __device__ __forceinline__ void wait_async() {
#ifdef HAVE_WAIT_BUILTIN
    __builtin_amdgcn_s_wait_asynccnt(0);
#else
    asm volatile("s_wait_asynccnt 0x0" ::: "memory");
#endif
}

// ---------------------------------------------------------------------------
// bf16 WMMA GEMM:  C[M,N] = act(A[M,K] * B[K,N] + bias)
//   A   : bf16 row-major (lda, optional batch stride), fragments read directly
//   Bt  : bf16 TRANSPOSED weights: Bt[n*ldb + k]  (K pairs contiguous)
//   Cbf : optional bf16 out, row-major ldc w/ column offset ccol0, or
//         transposed (Cbf[col*ldc + row]) when transC != 0
//   Cf  : optional fp32 out, row-major ldcf
//
// Block = 256 threads = 8 waves. Each wave owns a 32x64 strip (2 M-tiles x
// 4 N-tiles, 8 f32x8 accumulators). The 64x32 B tile for each k-step is
// staged once per block into LDS in fragment order via async loads
// (double buffered), then each wave reads it back as aligned ds_load_b128.
// grid = (M/256, N/64, batches).  K must be a multiple of 32.
// ---------------------------------------------------------------------------
__global__ __launch_bounds__(256)
void wmma_gemm_bf16(const unsigned short* __restrict__ A, int lda, long aBatch,
                    const unsigned short* __restrict__ Bt, int ldb, long bBatch,
                    const float* __restrict__ bias,
                    unsigned short* __restrict__ Cbf, int ldc, long cBatch, int ccol0,
                    float* __restrict__ Cf, int ldcf,
                    int K, int doRelu, int transC)
{
    __shared__ unsigned short sB[2][2048];   // 2 x 4KB: [t(4)][n(16)][hi(2)][16 bf16]

    const int tid  = threadIdx.x;
    const int lane = tid & 31;
    const int wave = tid >> 5;
    const int m16  = lane & 15;      // row within A tile / col within B,C tile
    const int hi   = lane >> 4;      // lane-group select
    const int rowb = (blockIdx.x * 8 + wave) * 32;
    const int colb = blockIdx.y * 64;

    A  += (long)blockIdx.z * aBatch;
    Bt += (long)blockIdx.z * bBatch;

    // Staging chunk for this thread: 16 bytes of the 4KB B tile.
    // chunk c = run*2 + half ; run = t*32 + n*2 + hi  (fragment order)
    const int s_half = tid & 1;
    const int s_run  = tid >> 1;
    const int s_t    = s_run >> 5;
    const int s_n    = (s_run >> 1) & 15;
    const int s_hi   = s_run & 1;
    const unsigned short* s_src =
        Bt + (long)(colb + s_t * 16 + s_n) * ldb + s_hi * 8 + s_half * 16;
    unsigned short* s_dst0 = &sB[0][tid * 8];
    unsigned short* s_dst1 = &sB[1][tid * 8];

    f32x8 acc[2][4];
#pragma unroll
    for (int mt = 0; mt < 2; ++mt)
#pragma unroll
        for (int t = 0; t < 4; ++t)
            acc[mt][t] = (f32x8){0.f,0.f,0.f,0.f,0.f,0.f,0.f,0.f};

    const unsigned short* arow = A + (long)(rowb + m16) * lda;

    // Prologue: stage k-step 0 into buffer 0.
    async_copy16(s_src, s_dst0);

    int buf = 0;
    for (int k0 = 0; k0 < K; k0 += 32, buf ^= 1) {
        wait_async();          // our async writes into sB[buf] are done
        __syncthreads();       // everyone's are, and sB[buf^1] is free again

        if (k0 + 32 < K)       // stage next k-step into the other buffer
            async_copy16(s_src + (k0 + 32), buf ? s_dst0 : s_dst1);

        // A fragments (16x32 bf16 each): VGPR j holds K pair
        //   j<4 : K = hi*8 + 2j ; j>=4: K = 16 + hi*8 + 2(j-4)
        FragBF a0, a1;
#pragma unroll
        for (int j = 0; j < 8; ++j) {
            int kk = k0 + ((j >> 2) << 4) + hi * 8 + ((j & 3) << 1);
            a0.u[j] = *(const unsigned int*)(arow + kk);
            a1.u[j] = *(const unsigned int*)(arow + (long)16 * lda + kk);
        }
#pragma unroll
        for (int t = 0; t < 4; ++t) {
            // B fragment from LDS, fragment order => 2x ds_load_b128.
            FragBF b;
            const unsigned int* bp =
                (const unsigned int*)&sB[buf][(t * 32 + m16 * 2 + hi) * 16];
#pragma unroll
            for (int j = 0; j < 8; ++j) b.u[j] = bp[j];

            acc[0][t] = __builtin_amdgcn_wmma_f32_16x16x32_bf16(
                false, a0.v, false, b.v, (short)0, acc[0][t], false, false);
            acc[1][t] = __builtin_amdgcn_wmma_f32_16x16x32_bf16(
                false, a1.v, false, b.v, (short)0, acc[1][t], false, false);
        }
    }

    // Epilogue: bias + relu + store (bf16 and/or fp32).
#pragma unroll
    for (int mt = 0; mt < 2; ++mt) {
#pragma unroll
        for (int t = 0; t < 4; ++t) {
            const int col = colb + t * 16 + m16;
            const float bv = bias ? bias[col] : 0.f;
#pragma unroll
            for (int r = 0; r < 8; ++r) {
                const int row = rowb + mt * 16 + hi * 8 + r;  // C VGPR r -> M=r+8*hi
                float v = acc[mt][t][r] + bv;
                if (doRelu) v = v > 0.f ? v : 0.f;
                if (Cbf) {
                    if (transC)
                        Cbf[(long)blockIdx.z * cBatch + (long)col * ldc + row] = f2bf(v);
                    else
                        Cbf[(long)blockIdx.z * cBatch + (long)row * ldc + ccol0 + col] = f2bf(v);
                }
                if (Cf)
                    Cf[(long)row * ldcf + col] = v;
            }
        }
    }
}

// ---------------------------------------------------------------------------
// Weight conversion: fp32 W[K,N] -> bf16 transposed Wt[N, Kpad], zero-padded K.
// ---------------------------------------------------------------------------
__global__ void conv_wT(const float* __restrict__ W, unsigned short* __restrict__ Wt,
                        int K, int N, int Kpad)
{
    long i = (long)blockIdx.x * 256 + threadIdx.x;
    long total = (long)N * Kpad;
    if (i >= total) return;
    int n = (int)(i / Kpad), k = (int)(i % Kpad);
    Wt[i] = (k < K) ? f2bf(W[(long)k * N + n]) : (unsigned short)0;
}

// Elementwise fp32 -> bf16
__global__ void conv_bf16(const float* __restrict__ X, unsigned short* __restrict__ Y, long n)
{
    long i = (long)blockIdx.x * 256 + threadIdx.x;
    if (i < n) Y[i] = f2bf(X[i]);
}

// emb[t, 0:320] = sem[sc[t]] - sem[oc[t]]  (cols 300..319 zero-padded), bf16
__global__ void emb_kernel(const float* __restrict__ sem,
                           const int* __restrict__ sc, const int* __restrict__ oc,
                           unsigned short* __restrict__ emb)
{
    long i = (long)blockIdx.x * 256 + threadIdx.x;   // T*320 threads
    int t = (int)(i / 320), c = (int)(i % 320);
    float v = 0.f;
    if (c < 300) v = sem[(long)sc[t] * 300 + c] - sem[(long)oc[t] * 300 + c];
    emb[i] = f2bf(v);
}

// Location layer 1 (K=8 -> scalar math): hid[t,j] = relu(loc[t,:]@w1[:,j] + b1[j])
__global__ void loc1_kernel(const float* __restrict__ loc,
                            const float* __restrict__ w1, const float* __restrict__ b1,
                            unsigned short* __restrict__ hid)
{
    long i = (long)blockIdx.x * 256 + threadIdx.x;   // T*64 threads
    int t = (int)(i >> 6), j = (int)(i & 63);
    float s = b1[j];
#pragma unroll
    for (int k = 0; k < 8; ++k) s += loc[t * 8 + k] * w1[k * 64 + j];
    hid[i] = f2bf(s > 0.f ? s : 0.f);
}

// Softmax over last dim (128), one wave per row, bf16 out.
__global__ __launch_bounds__(256)
void softmax128(const float* __restrict__ att, unsigned short* __restrict__ out)
{
    const int lane = threadIdx.x & 31;
    const int wave = threadIdx.x >> 5;
    const long row = (long)blockIdx.x * 8 + wave;    // TT rows
    const float* x = att + row * 128;

    float v[4], m = -3.402823466e38f;
#pragma unroll
    for (int i = 0; i < 4; ++i) { v[i] = x[lane + 32 * i]; m = fmaxf(m, v[i]); }
#pragma unroll
    for (int off = 16; off > 0; off >>= 1) m = fmaxf(m, __shfl_xor(m, off, 32));

    float s = 0.f;
#pragma unroll
    for (int i = 0; i < 4; ++i) { v[i] = __expf(v[i] - m); s += v[i]; }
#pragma unroll
    for (int off = 16; off > 0; off >>= 1) s += __shfl_xor(s, off, 32);
    const float inv = 1.f / s;
#pragma unroll
    for (int i = 0; i < 4; ++i) out[row * 128 + lane + 32 * i] = f2bf(v[i] * inv);
}

// Final head: pred[t] = sigmoid(pair512b[t,:] . w3 + b3). One wave per row.
__global__ __launch_bounds__(256)
void head_kernel(const float* __restrict__ x, const float* __restrict__ w3,
                 const float* __restrict__ b3, float* __restrict__ pred)
{
    const int lane = threadIdx.x & 31;
    const int wave = threadIdx.x >> 5;
    const long t = (long)blockIdx.x * 8 + wave;
    float s = 0.f;
#pragma unroll
    for (int i = 0; i < 16; ++i) {
        int c = lane + 32 * i;
        s += x[t * 512 + c] * w3[c];
    }
#pragma unroll
    for (int off = 16; off > 0; off >>= 1) s += __shfl_xor(s, off, 32);
    if (lane == 0) {
        float z = s + b3[0];
        pred[t] = 1.f / (1.f + __expf(-z));
    }
}

// ---------------------------------------------------------------------------
// Host launcher
// ---------------------------------------------------------------------------
extern "C" void kernel_launch(void* const* d_in, const int* in_sizes, int n_in,
                              void* d_out, int out_size, void* d_ws, size_t ws_size,
                              hipStream_t stream)
{
    (void)in_sizes; (void)n_in; (void)out_size; (void)ws_size;

    // ---- inputs (setup_inputs dict order; params flattened in insertion order)
    const float* pair_features = (const float*)d_in[0];   // [T,2048]
    const float* pair_locs     = (const float*)d_in[1];   // [T,8]
    const int*   sub_cls       = (const int*)d_in[2];     // [T]
    const int*   obj_cls       = (const int*)d_in[3];     // [T]
    const float* attention     = (const float*)d_in[4];   // [B,P,N]
    const float* inst_feat     = (const float*)d_in[5];   // [B,N,512]
    const float* sem     = (const float*)d_in[6];
    const float* lang_w1 = (const float*)d_in[7];  const float* lang_b1 = (const float*)d_in[8];
    const float* lang_w2 = (const float*)d_in[9];  const float* lang_b2 = (const float*)d_in[10];
    const float* loc_w1  = (const float*)d_in[11]; const float* loc_b1  = (const float*)d_in[12];
    const float* loc_w2  = (const float*)d_in[13]; const float* loc_b2  = (const float*)d_in[14];
    const float* vis_w1  = (const float*)d_in[15]; const float* vis_b1  = (const float*)d_in[16];
    const float* vis_w2  = (const float*)d_in[17]; const float* vis_b2  = (const float*)d_in[18];
    const float* pair_w1 = (const float*)d_in[19]; const float* pair_b1 = (const float*)d_in[20];
    const float* gcn_w1  = (const float*)d_in[21]; const float* gcn_b1  = (const float*)d_in[22];
    const float* gcn_w2  = (const float*)d_in[23]; const float* gcn_b2  = (const float*)d_in[24];
    const float* pair_w2 = (const float*)d_in[25]; const float* pair_b2 = (const float*)d_in[26];
    const float* pair_w3 = (const float*)d_in[27]; const float* pair_b3 = (const float*)d_in[28];

    float* pred     = (float*)d_out;             // [T]
    float* pair512b = (float*)d_out + TT;        // [T,512]

    // ---- workspace bump allocator (bf16 buffers)
    size_t off = 0;
    auto alloc = [&](size_t elems) -> unsigned short* {
        unsigned short* p = (unsigned short*)((char*)d_ws + off);
        off += (elems * 2 + 255) & ~(size_t)255;
        return p;
    };
    unsigned short* wt_lang1 = alloc(512  * 320);   // [N=512, Kpad=320]
    unsigned short* wt_lang2 = alloc(512  * 512);
    unsigned short* wt_loc2  = alloc(512  * 64);
    unsigned short* wt_vis1  = alloc(1024 * 2048);
    unsigned short* wt_vis2  = alloc(512  * 1024);
    unsigned short* wt_pair1 = alloc(512  * 1536);
    unsigned short* wt_gcn1  = alloc(256  * 512);
    unsigned short* wt_gcn2  = alloc(512  * 256);
    unsigned short* wt_pair2 = alloc(512  * 1024);
    unsigned short* emb      = alloc((size_t)TT * 320);
    unsigned short* pf_bf    = alloc((size_t)TT * 2048);
    unsigned short* hid      = alloc((size_t)TT * 1024);  // reused per branch
    unsigned short* feat     = alloc((size_t)TT * 1536);  // lang|loc|vis concat
    unsigned short* upd      = alloc((size_t)TT * 1024);  // pair512|gcn concat
    unsigned short* asoft    = alloc((size_t)TT * 128);
    unsigned short* instf_bf = alloc((size_t)TB * TN * 512);
    unsigned short* ht       = alloc(256 * (TB * TN));    // h transposed [256,1024]
    unsigned short* agg      = alloc((size_t)TT * 256);

    const dim3 blk(256);
    auto cdiv = [](long a, long b) { return (unsigned)((a + b - 1) / b); };

    // ---- weight conversions (fp32 -> transposed bf16, K zero-padded)
    conv_wT<<<cdiv(512L*320,256),  blk, 0, stream>>>(lang_w1, wt_lang1, 300, 512, 320);
    conv_wT<<<cdiv(512L*512,256),  blk, 0, stream>>>(lang_w2, wt_lang2, 512, 512, 512);
    conv_wT<<<cdiv(512L*64,256),   blk, 0, stream>>>(loc_w2,  wt_loc2,   64, 512, 64);
    conv_wT<<<cdiv(1024L*2048,256),blk, 0, stream>>>(vis_w1,  wt_vis1, 2048,1024, 2048);
    conv_wT<<<cdiv(512L*1024,256), blk, 0, stream>>>(vis_w2,  wt_vis2, 1024, 512, 1024);
    conv_wT<<<cdiv(512L*1536,256), blk, 0, stream>>>(pair_w1, wt_pair1,1536, 512, 1536);
    conv_wT<<<cdiv(256L*512,256),  blk, 0, stream>>>(gcn_w1,  wt_gcn1,  512, 256, 512);
    conv_wT<<<cdiv(512L*256,256),  blk, 0, stream>>>(gcn_w2,  wt_gcn2,  256, 512, 256);
    conv_wT<<<cdiv(512L*1024,256), blk, 0, stream>>>(pair_w2, wt_pair2,1024, 512, 1024);

    // ---- input conversions
    emb_kernel<<<cdiv((long)TT*320,256), blk, 0, stream>>>(sem, sub_cls, obj_cls, emb);
    conv_bf16<<<cdiv((long)TT*2048,256), blk, 0, stream>>>(pair_features, pf_bf, (long)TT*2048);
    conv_bf16<<<cdiv((long)TB*TN*512,256), blk, 0, stream>>>(inst_feat, instf_bf, (long)TB*TN*512);

    // ---- language branch: [T,320]x[320,512] -> relu -> [T,512]x[512,512] -> feat[:,0:512]
    wmma_gemm_bf16<<<dim3(TT/256, 512/64, 1), blk, 0, stream>>>(
        emb, 320, 0, wt_lang1, 320, 0, lang_b1, hid, 512, 0, 0, nullptr, 0, 320, 1, 0);
    wmma_gemm_bf16<<<dim3(TT/256, 512/64, 1), blk, 0, stream>>>(
        hid, 512, 0, wt_lang2, 512, 0, lang_b2, feat, 1536, 0, 0, nullptr, 0, 512, 1, 0);

    // ---- location branch: scalar K=8 layer, then [T,64]x[64,512] -> feat[:,512:1024]
    loc1_kernel<<<cdiv((long)TT*64,256), blk, 0, stream>>>(pair_locs, loc_w1, loc_b1, hid);
    wmma_gemm_bf16<<<dim3(TT/256, 512/64, 1), blk, 0, stream>>>(
        hid, 64, 0, wt_loc2, 64, 0, loc_b2, feat, 1536, 0, 512, nullptr, 0, 64, 1, 0);

    // ---- visual branch: [T,2048]x[2048,1024] -> [T,1024]x[1024,512] -> feat[:,1024:1536]
    wmma_gemm_bf16<<<dim3(TT/256, 1024/64, 1), blk, 0, stream>>>(
        pf_bf, 2048, 0, wt_vis1, 2048, 0, vis_b1, hid, 1024, 0, 0, nullptr, 0, 2048, 1, 0);
    wmma_gemm_bf16<<<dim3(TT/256, 512/64, 1), blk, 0, stream>>>(
        hid, 1024, 0, wt_vis2, 1024, 0, vis_b2, feat, 1536, 0, 1024, nullptr, 0, 1024, 1, 0);

    // ---- pair512 = relu(feat @ pair_w1 + b) -> upd[:,0:512]
    wmma_gemm_bf16<<<dim3(TT/256, 512/64, 1), blk, 0, stream>>>(
        feat, 1536, 0, wt_pair1, 1536, 0, pair_b1, upd, 1024, 0, 0, nullptr, 0, 1536, 1, 0);

    // ---- GCN: h = relu(instf @ gcn_w1 + b), stored transposed ht[256,1024]
    wmma_gemm_bf16<<<dim3((TB*TN)/256, 256/64, 1), blk, 0, stream>>>(
        instf_bf, 512, 0, wt_gcn1, 512, 0, gcn_b1, ht, TB*TN, 0, 0, nullptr, 0, 512, 1, 1);

    // ---- softmax(attention) -> asoft bf16 [T,128]
    softmax128<<<TT/8, blk, 0, stream>>>(attention, asoft);

    // ---- agg[b] = asoft[b] @ h[b] : batched [4096,128]x[128,256], B via ht strides
    wmma_gemm_bf16<<<dim3(TP/256, 256/64, TB), blk, 0, stream>>>(
        asoft, 128, (long)TP*128, ht, TB*TN, 128, nullptr,
        agg, 256, (long)TP*256, 0, nullptr, 0, 128, 0, 0);

    // ---- gcn = relu(agg @ gcn_w2 + b) -> upd[:,512:1024]
    wmma_gemm_bf16<<<dim3(TT/256, 512/64, 1), blk, 0, stream>>>(
        agg, 256, 0, wt_gcn2, 256, 0, gcn_b2, upd, 1024, 0, 512, nullptr, 0, 256, 1, 0);

    // ---- pair512b = relu(upd @ pair_w2 + b) -> fp32 directly into d_out[T:]
    wmma_gemm_bf16<<<dim3(TT/256, 512/64, 1), blk, 0, stream>>>(
        upd, 1024, 0, wt_pair2, 1024, 0, pair_b2, nullptr, 0, 0, 0,
        pair512b, 512, 1024, 1, 0);

    // ---- pred = sigmoid(pair512b . w3 + b3)
    head_kernel<<<TT/8, blk, 0, stream>>>(pair512b, pair_w3, pair_b3, pred);
}
